// StackLSTM_26895085207723
// MI455X (gfx1250) — matrix operand order repaired and checked by reading, not compiled, not run on hardware
//
#include <hip/hip_runtime.h>
#include <cstdint>

typedef __attribute__((ext_vector_type(16))) __bf16 v16bf;
typedef __attribute__((ext_vector_type(8)))  float  v8f;

#define T_STEPS 256
#define B_DIM   64
#define H_DIM   1024
#define NT      256      // N tiles (4096 / 16)
#define KT      32       // K tiles per 1024-wide operand (1024 / 32)
#define FRAG_U16 512     // bf16 elements per B fragment (32 lanes * 16)
#define BH      (B_DIM * H_DIM)

#if __has_builtin(__builtin_amdgcn_sched_group_barrier)
// masks: 0x2 = VALU, 0x8 = MFMA/WMMA, 0x20 = VMEM read
#define SCHED_GROUP(mask, n) __builtin_amdgcn_sched_group_barrier((mask), (n), 0)
#else
#define SCHED_GROUP(mask, n)
#endif

__device__ __forceinline__ float sigf(float x)  { return 1.0f / (1.0f + __expf(-x)); }
__device__ __forceinline__ float tanhf_(float x){ return 2.0f / (1.0f + __expf(-2.0f * x)) - 1.0f; }

// --------------------------------------------------------------------------
// Convert a [4096 x 1024] row-major fp32 weight matrix (torch W.T convention:
// gates[b,n] = sum_k x[b,k] * W[n,k]) into bf16 fragments in the CDNA5 WMMA
// 16-bit B-matrix (32x16, K x N) lane order:
//   lane L: N = tile_n*16 + (L&15);  K = tile_k*32 + (L<16 ? 0 : 8) + e + (e<8 ? 0 : 8)
//   fragment f = ntile*KT + ktile, 32 lanes x 16 bf16 contiguous per lane.
// Hot loop then reads each lane's 32B contiguously -> fully coalesced b128s.
// --------------------------------------------------------------------------
__global__ __launch_bounds__(256) void wconvert(const float* __restrict__ W,
                                                unsigned short* __restrict__ out) {
  int tid  = blockIdx.x * 256 + threadIdx.x;      // NT*KT*32 = 262144 threads
  int lane = tid & 31;
  int f    = tid >> 5;                             // fragment id = nt*KT + kt
  int kt   = f & (KT - 1);
  int nt   = f >> 5;                               // KT == 32
  int n     = nt * 16 + (lane & 15);
  int kbase = kt * 32 + ((lane & 16) ? 8 : 0);
  const float* src    = W + (size_t)n * 1024 + kbase;
  unsigned short* dst = out + (size_t)f * FRAG_U16 + lane * 16;
#pragma unroll
  for (int e = 0; e < 16; ++e) {
    __bf16 b = (__bf16)src[e + (e & 8)];
    dst[e] = __builtin_bit_cast(unsigned short, b);
  }
}

__global__ __launch_bounds__(256) void copy_f32(const float* __restrict__ s,
                                                float* __restrict__ d, int n) {
  int i = blockIdx.x * 256 + threadIdx.x;
  if (i < n) d[i] = s[i];
}

// Raw fp32 A-tile data for one lane (4 x float4 = 16 values), and its
// conversion into the CDNA5 16-bit A-matrix (16x32) fragment layout.
// Lane L holds row M = (L&15); with ko = (L<16 ? 0 : 8):
//   halves 0..7  -> K = ko + 0..7 ; halves 8..15 -> K = ko + 16..23
struct ARaw { float4 a, b, c, d; };

__device__ __forceinline__ ARaw load_a_raw(const float* __restrict__ p) {
  ARaw r;
  r.a = *(const float4*)(p);
  r.b = *(const float4*)(p + 4);
  r.c = *(const float4*)(p + 16);
  r.d = *(const float4*)(p + 20);
  return r;
}

__device__ __forceinline__ v16bf cvt_a_frag(const ARaw& r) {
  v16bf v;
  v[0]  = (__bf16)r.a.x; v[1]  = (__bf16)r.a.y; v[2]  = (__bf16)r.a.z; v[3]  = (__bf16)r.a.w;
  v[4]  = (__bf16)r.b.x; v[5]  = (__bf16)r.b.y; v[6]  = (__bf16)r.b.z; v[7]  = (__bf16)r.b.w;
  v[8]  = (__bf16)r.c.x; v[9]  = (__bf16)r.c.y; v[10] = (__bf16)r.c.z; v[11] = (__bf16)r.c.w;
  v[12] = (__bf16)r.d.x; v[13] = (__bf16)r.d.y; v[14] = (__bf16)r.d.z; v[15] = (__bf16)r.d.w;
  return v;
}

// --------------------------------------------------------------------------
// One fused LSTM layer-step: gates = [x|h] @ [Wih|Whh]^T + b_ih + b_hh,
// then i,f,g,o activations and the cell update, all in-register.
// Grid: 32 blocks x 256 threads = 256 waves. Wave gw: mt = gw&3 (M tile of
// 64-batch), ng = gw>>2 (16-wide slice of H). The wave accumulates the four
// gate tiles at N = g*1024 + ng*16 so the nonlinearity needs no cross-wave
// exchange. h is double-buffered across steps; c updated in place.
//
// The K loop is software-pipelined and the schedule is pinned with
// sched_group_barrier into (VMEM x12)(VALU x8)(WMMA x4) groups per k-tile:
//   VMEM: raw A loads for kt+2 and B fragment loads for kt+1
//   VALU: bf16 convert of A for kt+1 (loaded one iteration ago)
//   WMMA: 4 gate WMMAs for kt, whose loads completed a full iteration ago.
// This forces double-buffered registers (no s_wait_loadcnt 0 load-use
// stalls) and puts >=16 instructions between a WMMA's A-read and the cvt
// that overwrites it (no WAR hazard v_nops).
// --------------------------------------------------------------------------
__global__ __launch_bounds__(256) void lstm_step(
    const float* __restrict__ x,          // [64][1024] layer input (fp32)
    const float* __restrict__ h_prev,     // [64][1024]
    float* __restrict__ h_out,            // [64][1024]
    float* __restrict__ c,                // [64][1024] in-place
    const unsigned short* __restrict__ Wih,  // bf16 fragments
    const unsigned short* __restrict__ Whh,  // bf16 fragments
    const float* __restrict__ b_ih,       // [4096]
    const float* __restrict__ b_hh)       // [4096]
{
  const int lane = threadIdx.x & 31;
  const int wid  = threadIdx.x >> 5;
  const int gw   = blockIdx.x * 8 + wid;  // 0..255
  const int mt   = gw & 3;                // M tile (batch / 16)
  const int ng   = gw >> 2;               // 0..63, 16-wide H slice

  const int mrow = mt * 16 + (lane & 15);
  const int ko   = (lane & 16) ? 8 : 0;

  v8f acc0 = {}, acc1 = {}, acc2 = {}, acc3 = {};

  for (int half = 0; half < 2; ++half) {
    const float* __restrict__ S = half ? h_prev : x;
    const unsigned short* __restrict__ Wu = half ? Whh : Wih;
    const float* __restrict__ rowp = S + (size_t)mrow * 1024 + ko;
    // Per-gate fragment base pointers (v16bf units): ((g*64+ng)*KT)*32 + lane
    const v16bf* __restrict__ Wf = reinterpret_cast<const v16bf*>(Wu);
    const v16bf* __restrict__ p0 = Wf + ((size_t)((0 * 64 + ng) * KT) << 5) + lane;
    const v16bf* __restrict__ p1 = Wf + ((size_t)((1 * 64 + ng) * KT) << 5) + lane;
    const v16bf* __restrict__ p2 = Wf + ((size_t)((2 * 64 + ng) * KT) << 5) + lane;
    const v16bf* __restrict__ p3 = Wf + ((size_t)((3 * 64 + ng) * KT) << 5) + lane;

    // ---- pipeline prologue ----
    ARaw  ar = load_a_raw(rowp);                 // A raw, kt=0
    v16bf af = cvt_a_frag(ar);                   // A bf16, kt=0
    ARaw  arn = load_a_raw(rowp + 32);           // A raw, kt=1
    v16bf b0 = p0[0], b1 = p1[0], b2 = p2[0], b3 = p3[0];   // B, kt=0

#pragma unroll 2
    for (int kt = 0; kt < KT - 2; ++kt) {
      // VMEM group: A raw for kt+2, B fragments for kt+1.
      ARaw  arn2 = load_a_raw(rowp + (kt + 2) * 32);
      v16bf c0 = p0[(size_t)(kt + 1) << 5];
      v16bf c1 = p1[(size_t)(kt + 1) << 5];
      v16bf c2 = p2[(size_t)(kt + 1) << 5];
      v16bf c3 = p3[(size_t)(kt + 1) << 5];
      // VALU group: convert A for kt+1 (raw data loaded last iteration).
      v16bf afn = cvt_a_frag(arn);
      // WMMA group: consume kt.
      acc0 = __builtin_amdgcn_wmma_f32_16x16x32_bf16(false, af, false, b0, (short)0, acc0, false, false);
      acc1 = __builtin_amdgcn_wmma_f32_16x16x32_bf16(false, af, false, b1, (short)0, acc1, false, false);
      acc2 = __builtin_amdgcn_wmma_f32_16x16x32_bf16(false, af, false, b2, (short)0, acc2, false, false);
      acc3 = __builtin_amdgcn_wmma_f32_16x16x32_bf16(false, af, false, b3, (short)0, acc3, false, false);
      af = afn;  ar = arn;  arn = arn2;
      b0 = c0;  b1 = c1;  b2 = c2;  b3 = c3;
      SCHED_GROUP(0x20, 12);   // 12 VMEM reads
      SCHED_GROUP(0x2,   8);   // 8 VALU (bf16 converts)
      SCHED_GROUP(0x8,   4);   // 4 WMMA
    }
    {   // kt = KT-2: B for KT-1, convert A for KT-1, WMMAs for KT-2.
      v16bf c0 = p0[(size_t)(KT - 1) << 5];
      v16bf c1 = p1[(size_t)(KT - 1) << 5];
      v16bf c2 = p2[(size_t)(KT - 1) << 5];
      v16bf c3 = p3[(size_t)(KT - 1) << 5];
      v16bf afn = cvt_a_frag(arn);
      acc0 = __builtin_amdgcn_wmma_f32_16x16x32_bf16(false, af, false, b0, (short)0, acc0, false, false);
      acc1 = __builtin_amdgcn_wmma_f32_16x16x32_bf16(false, af, false, b1, (short)0, acc1, false, false);
      acc2 = __builtin_amdgcn_wmma_f32_16x16x32_bf16(false, af, false, b2, (short)0, acc2, false, false);
      acc3 = __builtin_amdgcn_wmma_f32_16x16x32_bf16(false, af, false, b3, (short)0, acc3, false, false);
      af = afn;  b0 = c0;  b1 = c1;  b2 = c2;  b3 = c3;
    }
    // kt = KT-1: everything already resident.
    acc0 = __builtin_amdgcn_wmma_f32_16x16x32_bf16(false, af, false, b0, (short)0, acc0, false, false);
    acc1 = __builtin_amdgcn_wmma_f32_16x16x32_bf16(false, af, false, b1, (short)0, acc1, false, false);
    acc2 = __builtin_amdgcn_wmma_f32_16x16x32_bf16(false, af, false, b2, (short)0, acc2, false, false);
    acc3 = __builtin_amdgcn_wmma_f32_16x16x32_bf16(false, af, false, b3, (short)0, acc3, false, false);
  }

  // Epilogue: bias + activations + cell update.
  // C/D layout: VGPR r, lane L -> M = r + (L<16 ? 0 : 8), N = L&15.
  const int ncol  = ng * 16 + (lane & 15);
  const float bi0 = b_ih[ncol]        + b_hh[ncol];
  const float bi1 = b_ih[1024 + ncol] + b_hh[1024 + ncol];
  const float bi2 = b_ih[2048 + ncol] + b_hh[2048 + ncol];
  const float bi3 = b_ih[3072 + ncol] + b_hh[3072 + ncol];
  const int mbase = mt * 16 + ((lane & 16) ? 8 : 0);
#pragma unroll
  for (int r = 0; r < 8; ++r) {
    const int idx = (mbase + r) * 1024 + ncol;
    float iv = sigf(acc0[r] + bi0);
    float fv = sigf(acc1[r] + bi1);
    float gv = tanhf_(acc2[r] + bi2);
    float ov = sigf(acc3[r] + bi3);
    float cn = fv * c[idx] + iv * gv;
    c[idx]     = cn;
    h_out[idx] = ov * tanhf_(cn);
  }
}

extern "C" void kernel_launch(void* const* d_in, const int* in_sizes, int n_in,
                              void* d_out, int out_size, void* d_ws, size_t ws_size,
                              hipStream_t stream) {
  const float* input_seq = (const float*)d_in[0];
  const float* W_ih_0    = (const float*)d_in[1];
  const float* W_hh_0    = (const float*)d_in[2];
  const float* b_ih_0    = (const float*)d_in[3];
  const float* b_hh_0    = (const float*)d_in[4];
  const float* W_ih_1    = (const float*)d_in[5];
  const float* W_hh_1    = (const float*)d_in[6];
  const float* b_ih_1    = (const float*)d_in[7];
  const float* b_hh_1    = (const float*)d_in[8];
  const float* h0_in     = (const float*)d_in[9];   // [2][64][1024]
  const float* c0_in     = (const float*)d_in[10];  // [2][64][1024]

  char* ws = (char*)d_ws;
  const size_t WB = (size_t)NT * KT * FRAG_U16 * sizeof(unsigned short); // 8 MB
  unsigned short* fW0i = (unsigned short*)(ws + 0 * WB);
  unsigned short* fW0h = (unsigned short*)(ws + 1 * WB);
  unsigned short* fW1i = (unsigned short*)(ws + 2 * WB);
  unsigned short* fW1h = (unsigned short*)(ws + 3 * WB);
  char* st = ws + 4 * WB;
  const size_t SB = (size_t)BH * sizeof(float);     // 256 KB
  float* h0buf[2] = { (float*)(st + 0 * SB), (float*)(st + 1 * SB) };
  float* h1buf[2] = { (float*)(st + 2 * SB), (float*)(st + 3 * SB) };
  float* c0buf    =   (float*)(st + 4 * SB);
  float* c1buf    =   (float*)(st + 5 * SB);

  // Weight convert + swizzle (deterministic, every call).
  const int cvtBlocks = (NT * KT * 32) / 256;       // 1024
  wconvert<<<cvtBlocks, 256, 0, stream>>>(W_ih_0, fW0i);
  wconvert<<<cvtBlocks, 256, 0, stream>>>(W_hh_0, fW0h);
  wconvert<<<cvtBlocks, 256, 0, stream>>>(W_ih_1, fW1i);
  wconvert<<<cvtBlocks, 256, 0, stream>>>(W_hh_1, fW1h);

  // State init.
  const int cpBlocks = BH / 256;
  copy_f32<<<cpBlocks, 256, 0, stream>>>(h0_in,       h0buf[0], BH);
  copy_f32<<<cpBlocks, 256, 0, stream>>>(h0_in + BH,  h1buf[0], BH);
  copy_f32<<<cpBlocks, 256, 0, stream>>>(c0_in,       c0buf,    BH);
  copy_f32<<<cpBlocks, 256, 0, stream>>>(c0_in + BH,  c1buf,    BH);

  // Sequential recurrence: two fused layer-step kernels per time step.
  for (int t = 0; t < T_STEPS; ++t) {
    const int rp = t & 1, wp = rp ^ 1;
    const float* x0 = input_seq + (size_t)t * BH;   // [64][1024] slice
    lstm_step<<<32, 256, 0, stream>>>(x0, h0buf[rp], h0buf[wp], c0buf,
                                      fW0i, fW0h, b_ih_0, b_hh_0);
    lstm_step<<<32, 256, 0, stream>>>(h0buf[wp], h1buf[rp], h1buf[wp], c1buf,
                                      fW1i, fW1h, b_ih_1, b_hh_1);
  }

  // After t = 255 (odd), the last write went to buffer 0.
  hipMemcpyAsync(d_out, h1buf[0], (size_t)BH * sizeof(float),
                 hipMemcpyDeviceToDevice, stream);
}
